// GRUEncoder_56547539419565
// MI455X (gfx1250) — matrix operand order repaired and checked by reading, not compiled, not run on hardware
//
#include <hip/hip_runtime.h>
#include <hip/hip_bf16.h>

typedef __attribute__((ext_vector_type(16))) __bf16 v16bf;
typedef __attribute__((ext_vector_type(8)))  float  v8f;

union ABOp { uint4 u[2]; v16bf v; };

#define LN_EPS 1e-5f

__device__ __forceinline__ float sigmoidf_(float x) { return 1.0f / (1.0f + __expf(-x)); }

// Issue 6 async b128 copies per thread: 1024 threads x 16B x 6 = 98304 B
// (one (64 x 768) bf16 Gi timestep slice) from global into an LDS buffer.
// Tracked by ASYNCcnt (6 increments per wave).
__device__ __forceinline__ void async_copy_gi(const __bf16* __restrict__ gsrc,
                                              char* lbuf, int tid) {
    unsigned lds0 = (unsigned)(uintptr_t)lbuf + tid * 16;              // addr[31:0] == LDS offset
    unsigned long long g0 = (unsigned long long)(uintptr_t)gsrc + tid * 16;
#pragma unroll
    for (int i = 0; i < 6; ++i) {
        unsigned lds = lds0 + i * 16384;
        unsigned long long g = g0 + i * 16384;
        asm volatile("global_load_async_to_lds_b128 %0, %1, off"
                     :: "v"(lds), "v"(g) : "memory");
    }
}

// ---------------------------------------------------------------------------
// f32 -> bf16 conversion
// ---------------------------------------------------------------------------
__global__ void f32_to_bf16_kernel(const float* __restrict__ s, __bf16* __restrict__ d, int n) {
    int i = blockIdx.x * 256 + threadIdx.x;
    if (i < n) d[i] = (__bf16)s[i];
}

// ---------------------------------------------------------------------------
// LayerNorm over C=256, input x is (B,T,C) f32, output xn is (T*B, C) bf16
// (row t*64+b). One wave per row, 8 rows per 256-thread block.
// ---------------------------------------------------------------------------
__global__ void ln_x_kernel(const float* __restrict__ x, const float* __restrict__ g,
                            const float* __restrict__ bb, __bf16* __restrict__ xn, int T) {
    int wave = threadIdx.x >> 5, lane = threadIdx.x & 31;
    long row = (long)blockIdx.x * 8 + wave;      // row = b*T + t
    int b = (int)(row / T), t = (int)(row % T);
    const float* xr = x + row * 256;
    float v[8], sum = 0.f, sq = 0.f;
#pragma unroll
    for (int i = 0; i < 8; ++i) {
        int c = lane + i * 32;
        v[i] = xr[c];
        sum += v[i]; sq += v[i] * v[i];
    }
#pragma unroll
    for (int m = 16; m >= 1; m >>= 1) { sum += __shfl_xor(sum, m, 32); sq += __shfl_xor(sq, m, 32); }
    float mean = sum * (1.0f / 256.0f);
    float var  = sq * (1.0f / 256.0f) - mean * mean;
    float inv  = rsqrtf(var + LN_EPS);
    __bf16* orow = xn + ((long)t * 64 + b) * 256;
#pragma unroll
    for (int i = 0; i < 8; ++i) {
        int c = lane + i * 32;
        orow[c] = (__bf16)((v[i] - mean) * inv * g[c] + bb[c]);
    }
}

// ---------------------------------------------------------------------------
// Generic WMMA GEMM:  Out(M,N) bf16 = A(M,K) bf16 @ Bw(N,K)^T bf16 + bias(N)
// One 16x16 output tile per wave, 8 waves per block, f32 accumulate.
// ---------------------------------------------------------------------------
__global__ void wmma_gemm_bias_kernel(const __bf16* __restrict__ A,
                                      const __bf16* __restrict__ Bw,
                                      const float* __restrict__ bias,
                                      __bf16* __restrict__ Out,
                                      int M, int N, int K) {
    int wave = threadIdx.x >> 5, lane = threadIdx.x & 31;
    int ntn = N >> 4;
    long total = (long)(M >> 4) * ntn;
    long tile = (long)blockIdx.x * 8 + wave;
    if (tile >= total) return;
    int mt = (int)(tile / ntn), nt = (int)(tile % ntn);
    int lm = lane & 15, hi = lane >> 4;

    const __bf16* arow = A  + (long)(mt * 16 + lm) * K;
    const __bf16* brow = Bw + (long)(nt * 16 + lm) * K;

    v8f acc = {};
#pragma unroll 2
    for (int kc = 0; kc < K; kc += 32) {
        ABOp a, b;
        a.u[0] = *(const uint4*)(arow + kc + hi * 8);
        a.u[1] = *(const uint4*)(arow + kc + 16 + hi * 8);
        b.u[0] = *(const uint4*)(brow + kc + hi * 8);
        b.u[1] = *(const uint4*)(brow + kc + 16 + hi * 8);
        acc = __builtin_amdgcn_wmma_f32_16x16x32_bf16(false, a.v, false, b.v,
                                                      (short)0, acc, false, false);
    }
    int ncol = nt * 16 + lm;
    float bv = bias ? bias[ncol] : 0.0f;
#pragma unroll
    for (int r = 0; r < 8; ++r) {
        long m = (long)mt * 16 + hi * 8 + r;
        Out[m * N + ncol] = (__bf16)(acc[r] + bv);
    }
}

// ---------------------------------------------------------------------------
// Persistent GRU scan. grid = 2 (dir 0 = forward, 1 = backward), 1024 threads.
// LDS: h (64x256 bf16, 32KB) + double-buffered Gi timestep slice (2x96KB),
// filled by global_load_async_to_lds_b128 one step ahead of consumption.
// Wave (wm, wc) owns rows wm*16..+16 and matched gate columns: c-tiles
// {2wc, 2wc+1} in each of the r/z/n segments -> the full GRU gate update
// happens on the WMMA accumulators in registers (no gh LDS round-trip).
// ---------------------------------------------------------------------------
__global__ void __launch_bounds__(1024)
gru_scan_kernel(const __bf16* __restrict__ Gi_f, const __bf16* __restrict__ Gi_b,
                const __bf16* __restrict__ Whh_f, const __bf16* __restrict__ Whh_b,
                const float* __restrict__ bhh_f, const float* __restrict__ bhh_b,
                __bf16* __restrict__ S, int T) {
    extern __shared__ char smem[];
    __bf16* h_lds  = (__bf16*)smem;                 // 64*256*2  = 32768 B
    char*  gi_buf0 = smem + 32768;                  // 64*768*2  = 98304 B
    char*  gi_buf1 = smem + 32768 + 98304;          // 64*768*2  = 98304 B

    const int dir = blockIdx.x;
    const __bf16* Gi  = dir ? Gi_b  : Gi_f;
    const __bf16* Whh = dir ? Whh_b : Whh_f;
    const float*  bhh = dir ? bhh_b : bhh_f;
    const int dir_off = dir ? 256 : 0;

    const int tid = threadIdx.x, wave = tid >> 5, lane = tid & 31;
    const int wm = wave & 3;          // M-tile (64 batch rows -> 4 tiles)
    const int wc = wave >> 2;         // [0,8): c-tiles {2wc, 2wc+1}
    const int lm = lane & 15, hi = lane >> 4;

    // hoisted recurrent biases for this lane's two columns
    float bR[2], bZ[2], bN[2];
#pragma unroll
    for (int u = 0; u < 2; ++u) {
        int c = (wc * 2 + u) * 16 + lm;
        bR[u] = bhh[c]; bZ[u] = bhh[256 + c]; bN[u] = bhh[512 + c];
    }

    // h0 = 0
    for (int i = tid; i < 64 * 256; i += 1024) h_lds[i] = (__bf16)0.0f;

    // prefetch the first timestep's Gi slice
    async_copy_gi(Gi + (long)(dir ? (T - 1) : 0) * 64 * 768, gi_buf0, tid);
    __syncthreads();

    for (int s = 0; s < T; ++s) {
        const int t = dir ? (T - 1 - s) : s;
        char* curbuf = (s & 1) ? gi_buf1 : gi_buf0;
        char* nxtbuf = (s & 1) ? gi_buf0 : gi_buf1;
        const bool more = (s + 1 < T);
        if (more) {
            int tn = dir ? (T - 2 - s) : (s + 1);
            async_copy_gi(Gi + (long)tn * 64 * 768, nxtbuf, tid);   // overlaps WMMAs
        }

        // ---- gh = h @ Whh^T : acc[u][seg] for {2 c-tiles} x {r,z,n} ----
        v8f acc[2][3];
#pragma unroll
        for (int u = 0; u < 2; ++u)
#pragma unroll
            for (int g = 0; g < 3; ++g) { v8f z = {}; acc[u][g] = z; }

        const __bf16* ar = h_lds + (wm * 16 + lm) * 256;
#pragma unroll 2
        for (int kc = 0; kc < 256; kc += 32) {
            ABOp a;
            a.u[0] = *(const uint4*)(ar + kc + hi * 8);
            a.u[1] = *(const uint4*)(ar + kc + 16 + hi * 8);
#pragma unroll
            for (int u = 0; u < 2; ++u) {
#pragma unroll
                for (int g = 0; g < 3; ++g) {
                    int n = g * 256 + (wc * 2 + u) * 16 + lm;
                    const __bf16* br = Whh + (long)n * 256 + kc + hi * 8;
                    ABOp b;
                    b.u[0] = *(const uint4*)br;
                    b.u[1] = *(const uint4*)(br + 16);
                    acc[u][g] = __builtin_amdgcn_wmma_f32_16x16x32_bf16(
                        false, a.v, false, b.v, (short)0, acc[u][g], false, false);
                }
            }
        }

        // current step's 6 async loads are the oldest outstanding; the 6 just
        // issued for step s+1 may remain in flight.
        if (more) asm volatile("s_wait_asynccnt 6" ::: "memory");
        else      asm volatile("s_wait_asynccnt 0" ::: "memory");
        __syncthreads();          // all h reads done + Gi(t) resident everywhere

        // ---- fused GRU gate update on accumulators ----
        const __bf16* gi = (const __bf16*)curbuf;
#pragma unroll
        for (int u = 0; u < 2; ++u) {
            int c = (wc * 2 + u) * 16 + lm;
#pragma unroll
            for (int r = 0; r < 8; ++r) {
                int m = wm * 16 + hi * 8 + r;
                float ir  = (float)gi[m * 768 + c];
                float iz  = (float)gi[m * 768 + 256 + c];
                float inn = (float)gi[m * 768 + 512 + c];
                float rg = sigmoidf_(ir + acc[u][0][r] + bR[u]);
                float zg = sigmoidf_(iz + acc[u][1][r] + bZ[u]);
                float ng = tanhf(inn + rg * (acc[u][2][r] + bN[u]));
                float hold = (float)h_lds[m * 256 + c];
                float hnew = (1.0f - zg) * ng + zg * hold;
                h_lds[m * 256 + c] = (__bf16)hnew;
                S[((long)t * 64 + m) * 512 + dir_off + c] = (__bf16)hnew;
            }
        }
        __syncthreads();          // h fully updated before next step's GEMM
    }
}

// ---------------------------------------------------------------------------
// Merge: sig = sigmoid(P), H = sig*sf + (1-sig)*sb, LayerNorm(H) -> out f32
// (B,T,C). One wave per row (row = t*64 + b), 8 rows per block.
// ---------------------------------------------------------------------------
__global__ void merge_ln_kernel(const __bf16* __restrict__ S, const __bf16* __restrict__ P,
                                const float* __restrict__ gH, const float* __restrict__ bH,
                                float* __restrict__ out, int T) {
    int wave = threadIdx.x >> 5, lane = threadIdx.x & 31;
    long row = (long)blockIdx.x * 8 + wave;      // row = t*64 + b
    int t = (int)(row >> 6), b = (int)(row & 63);
    float h[8], sum = 0.f, sq = 0.f;
#pragma unroll
    for (int i = 0; i < 8; ++i) {
        int c = lane + i * 32;
        float p  = (float)P[row * 256 + c];
        float sf = (float)S[row * 512 + c];
        float sb = (float)S[row * 512 + 256 + c];
        float sig = sigmoidf_(p);
        float v = sig * sf + (1.0f - sig) * sb;
        h[i] = v; sum += v; sq += v * v;
    }
#pragma unroll
    for (int m = 16; m >= 1; m >>= 1) { sum += __shfl_xor(sum, m, 32); sq += __shfl_xor(sq, m, 32); }
    float mean = sum * (1.0f / 256.0f);
    float var  = sq * (1.0f / 256.0f) - mean * mean;
    float inv  = rsqrtf(var + LN_EPS);
    float* orow = out + ((long)b * T + t) * 256;
#pragma unroll
    for (int i = 0; i < 8; ++i) {
        int c = lane + i * 32;
        orow[c] = (h[i] - mean) * inv * gH[c] + bH[c];
    }
}

// ---------------------------------------------------------------------------
extern "C" void kernel_launch(void* const* d_in, const int* in_sizes, int n_in,
                              void* d_out, int out_size, void* d_ws, size_t ws_size,
                              hipStream_t stream) {
    const int B = 64, T = 2048, C = 256;
    const float* x     = (const float*)d_in[0];
    // d_in[1] = ts (unused by reference forward)
    const float* Wih_f = (const float*)d_in[2];
    const float* Whh_f = (const float*)d_in[3];
    const float* bih_f = (const float*)d_in[4];
    const float* bhh_f = (const float*)d_in[5];
    const float* Wih_b = (const float*)d_in[6];
    const float* Whh_b = (const float*)d_in[7];
    const float* bih_b = (const float*)d_in[8];
    const float* bhh_b = (const float*)d_in[9];
    const float* Wg    = (const float*)d_in[10];
    const float* bg    = (const float*)d_in[11];
    const float* gx    = (const float*)d_in[12];
    const float* bx    = (const float*)d_in[13];
    const float* gH    = (const float*)d_in[14];
    const float* bH    = (const float*)d_in[15];
    float* out = (float*)d_out;

    const long M  = (long)B * T;                 // 131072 rows
    char* ws = (char*)d_ws;
    size_t off = 0;
    auto alloc = [&](size_t bytes) -> void* {
        void* p = ws + off; off += (bytes + 255) & ~(size_t)255; return p;
    };
    __bf16* xn    = (__bf16*)alloc((size_t)M * C * 2);          //  67 MB
    __bf16* Gi_f  = (__bf16*)alloc((size_t)M * 3 * C * 2);      // 201 MB
    __bf16* Gi_b  = (__bf16*)alloc((size_t)M * 3 * C * 2);      // 201 MB
    __bf16* S     = (__bf16*)alloc((size_t)M * 2 * C * 2);      // 134 MB
    __bf16* Wihf16 = (__bf16*)alloc((size_t)3 * C * C * 2);
    __bf16* Whhf16 = (__bf16*)alloc((size_t)3 * C * C * 2);
    __bf16* Wihb16 = (__bf16*)alloc((size_t)3 * C * C * 2);
    __bf16* Whhb16 = (__bf16*)alloc((size_t)3 * C * C * 2);
    __bf16* Wg16   = (__bf16*)alloc((size_t)C * 2 * C * 2);
    __bf16* P = Gi_f;   // alias: Gi_f is dead once the scans finish

    // 1) weight conversions to bf16
    const int nW = 3 * C * C;                    // 196608
    f32_to_bf16_kernel<<<(nW + 255) / 256, 256, 0, stream>>>(Wih_f, Wihf16, nW);
    f32_to_bf16_kernel<<<(nW + 255) / 256, 256, 0, stream>>>(Whh_f, Whhf16, nW);
    f32_to_bf16_kernel<<<(nW + 255) / 256, 256, 0, stream>>>(Wih_b, Wihb16, nW);
    f32_to_bf16_kernel<<<(nW + 255) / 256, 256, 0, stream>>>(Whh_b, Whhb16, nW);
    const int nWg = C * 2 * C;                   // 131072
    f32_to_bf16_kernel<<<(nWg + 255) / 256, 256, 0, stream>>>(Wg, Wg16, nWg);

    // 2) LayerNorm(x) -> xn (time-major bf16)
    ln_x_kernel<<<(int)(M / 8), 256, 0, stream>>>(x, gx, bx, xn, T);

    // 3) input projections Gi = xn @ Wih^T + bih    (M x 768, K=256)
    {
        long tiles = (M / 16) * (3 * C / 16);    // 393216
        int blocks = (int)((tiles + 7) / 8);
        wmma_gemm_bias_kernel<<<blocks, 256, 0, stream>>>(xn, Wihf16, bih_f, Gi_f,
                                                          (int)M, 3 * C, C);
        wmma_gemm_bias_kernel<<<blocks, 256, 0, stream>>>(xn, Wihb16, bih_b, Gi_b,
                                                          (int)M, 3 * C, C);
    }

    // 4) the two recurrent scans (persistent workgroups, 224KB LDS each)
    gru_scan_kernel<<<2, 1024, 32768 + 2 * 98304, stream>>>(
        Gi_f, Gi_b, Whhf16, Whhb16, bhh_f, bhh_b, S, T);

    // 5) merge pre-activation P = [sf|sb] @ Wg^T + bg   (M x 256, K=512)
    {
        long tiles = (M / 16) * (C / 16);        // 131072
        int blocks = (int)((tiles + 7) / 8);
        wmma_gemm_bias_kernel<<<blocks, 256, 0, stream>>>(S, Wg16, bg, P,
                                                          (int)M, C, 2 * C);
    }

    // 6) sigmoid blend + LayerNorm -> out (B,T,C) f32
    merge_ln_kernel<<<(int)(M / 8), 256, 0, stream>>>(S, P, gH, bH, out, T);
}
